// DenoisingNet_MLP_54623394070971
// MI455X (gfx1250) — matrix operand order repaired and checked by reading, not compiled, not run on hardware
//
#include <hip/hip_runtime.h>
#include <hip/hip_bf16.h>
#include <math.h>

typedef __attribute__((ext_vector_type(16))) _Float16 v16h;
typedef __attribute__((ext_vector_type(8)))  float    v8f;

#define NBATCH 4
#define PNUM   14641          // 121*121 stride-1 patches
#define PPAD   14656          // padded to multiple of 16 (916 tiles)
#define MALL   (NBATCH*PPAD)  // 58624 rows, 3664 tiles
#define MT_ALL 3664
#define MT_B   916

// ---------------------------------------------------------------- helpers
__device__ __forceinline__ float softf(float x, float l) {
  float s = (x > 0.f) ? 1.f : ((x < 0.f) ? -1.f : 0.f);
  return s * fmaxf(fabsf(x) - l, 0.f);
}

union AFrag { v16h v; unsigned u[8]; };

// Accumulate a 16x16 f32 tile: A row-major f16 [*, lda], B pre-packed per-lane.
__device__ __forceinline__ v8f wmma_rows(const _Float16* __restrict__ A, int lda, int m0,
                                         const _Float16* __restrict__ Bp_n, int KB, v8f acc) {
  const int lane = threadIdx.x & 31;
  const int half = lane >> 4;
  const _Float16* Arow = A + (size_t)(m0 + (lane & 15)) * lda;
  for (int kb = 0; kb < KB; ++kb) {
    AFrag a;
#pragma unroll
    for (int t = 0; t < 8; ++t) {
      // A-matrix 16x32 f16 layout: lanes 0-15 hold K pairs {0..7,16..23}, lanes 16-31 +8
      int kl = 2 * t + ((t >= 4) ? 8 : 0) + (half << 3);
      a.u[t] = *reinterpret_cast<const unsigned*>(Arow + kb * 32 + kl);
    }
    v16h b = *reinterpret_cast<const v16h*>(Bp_n + (((size_t)kb * 32 + lane) << 4));
    acc = __builtin_amdgcn_wmma_f32_16x16x32_f16(false, a.v, false, b, (short)0, acc,
                                                 false, false);
  }
  return acc;
}

// ---------------------------------------------------------------- packing
// Pack f32 B matrix [K x Nout] (strided) into WMMA per-lane order:
// dst[((nblk*KB+kblk)*32+lane)*16+e] = B[kblk*32 + (e&7) + 16*(e>>3) + 8*(lane>>4)][nblk*16 + (lane&15)]
__global__ void pack_b(const float* __restrict__ src, _Float16* __restrict__ dst,
                       int K, int Nout, int rowStride, int colStride,
                       long srcBatch, long dstBatch) {
  const int nb  = blockIdx.z;
  const int tid = blockIdx.x * blockDim.x + threadIdx.x;
  if (tid >= K * Nout) return;
  const int e    = tid & 15;
  const int lane = (tid >> 4) & 31;
  const int tile = tid >> 9;
  const int KB   = K >> 5;
  const int kblk = tile % KB;
  const int nblk = tile / KB;
  const int nn   = nblk * 16 + (lane & 15);
  const int kk   = kblk * 32 + (e & 7) + ((e >> 3) << 4) + ((lane >> 4) << 3);
  dst[(size_t)nb * dstBatch + tid] =
      (_Float16)src[(size_t)nb * srcBatch + (size_t)kk * rowStride + (size_t)nn * colStride];
}

// ---------------------------------------------------------------- im2col
__global__ void unfold_k(const float* __restrict__ x, _Float16* __restrict__ unf) {
  size_t tid = (size_t)blockIdx.x * blockDim.x + threadIdx.x;
  if (tid >= (size_t)MALL * 64) return;
  const int d = (int)(tid & 63);
  size_t rp   = tid >> 6;
  const int n = (int)(rp / PPAD);
  const int p = (int)(rp % PPAD);
  float v = 0.f;
  if (p < PNUM) {
    const int ho = p / 121, wo = p % 121;
    const int i = d >> 3, j = d & 7;
    v = x[((size_t)n << 14) + (size_t)(ho + i) * 128 + (wo + j)];
  }
  unf[tid] = (_Float16)v;
}

__global__ void ue_k(const float* __restrict__ x, _Float16* __restrict__ ue) {
  const int tid = blockIdx.x * blockDim.x + threadIdx.x; // 448*64
  if (tid >= 448 * 64) return;
  const int d = tid & 63;
  const int r = tid >> 6;
  const int n = r / 112, q = r % 112;
  const int p1 = 2 * q;                 // [::2][:112] of 16x16 stride-8 grid
  const int ho = p1 >> 4, wo = p1 & 15;
  const int i = d >> 3, j = d & 7;
  ue[tid] = (_Float16)x[((size_t)n << 14) + (size_t)(ho * 8 + i) * 128 + (wo * 8 + j)];
}

// ---------------------------------------------------------------- MLP GEMM
// mode: 0 = relu -> f16 | 1 = f32 | 2 = f32 * (1/c) | 3 = sigmoid -> f32
__global__ void gemm_mlp(const _Float16* __restrict__ A, const _Float16* __restrict__ Bp,
                         const float* __restrict__ bias, int Mtiles, int KB,
                         _Float16* __restrict__ outH, float* __restrict__ outF,
                         int ldOut, int colOffset, int mode,
                         const float* __restrict__ c_ptr) {
  const int mtile = blockIdx.y * (blockDim.x >> 5) + (threadIdx.x >> 5);
  if (mtile >= Mtiles) return;
  const int lane = threadIdx.x & 31;
  const int half = lane >> 4;
  const int n    = blockIdx.x * 16 + (lane & 15);
  const _Float16* Bp_n = Bp + (size_t)blockIdx.x * KB * 512;
  v8f acc;
  const float bv = bias[n];
#pragma unroll
  for (int r = 0; r < 8; ++r) acc[r] = bv;
  acc = wmma_rows(A, KB * 32, mtile * 16, Bp_n, KB, acc);
  const float inv_c = (mode == 2) ? (1.f / c_ptr[0]) : 1.f;
#pragma unroll
  for (int r = 0; r < 8; ++r) {
    const size_t m = (size_t)mtile * 16 + r + 8 * half;
    const float v = acc[r];
    if (mode == 0)      outH[m * ldOut + colOffset + n] = (_Float16)fmaxf(v, 0.f);
    else if (mode == 1) outF[m * ldOut + colOffset + n] = v;
    else if (mode == 2) outF[m * ldOut + colOffset + n] = v * inv_c;
    else                outF[m * ldOut + colOffset + n] = 1.f / (1.f + expf(-v));
  }
}

// y = unf @ Dc ; y_c = y/c ; z0 = soft(y, l)
__global__ void gemm_y(const _Float16* __restrict__ unf, const _Float16* __restrict__ Dcp,
                       const float* __restrict__ l, float* __restrict__ y_c,
                       _Float16* __restrict__ z0, const float* __restrict__ c_ptr) {
  const int mtile = blockIdx.y * (blockDim.x >> 5) + (threadIdx.x >> 5);
  if (mtile >= MT_B) return;
  const int lane = threadIdx.x & 31;
  const int half = lane >> 4;
  const int n    = blockIdx.x * 16 + (lane & 15);
  const int nb   = blockIdx.z;
  const _Float16* A    = unf + (size_t)nb * PPAD * 64;
  const _Float16* Bp_n = Dcp + (size_t)nb * 64 * 256 + (size_t)blockIdx.x * 1024;
  v8f acc = {};
  acc = wmma_rows(A, 64, mtile * 16, Bp_n, 2, acc);
  const float inv_c = 1.f / c_ptr[0];
#pragma unroll
  for (int r = 0; r < 8; ++r) {
    const size_t m   = (size_t)nb * PPAD + (size_t)mtile * 16 + r + 8 * half;
    const size_t idx = m * 256 + n;
    const float y = acc[r];
    y_c[idx] = y * inv_c;
    z0[idx]  = (_Float16)softf(y, l[idx]);
  }
}

// z_out = soft(z_in @ S + y_c, l)
__global__ void gemm_ista(const _Float16* __restrict__ zin, const _Float16* __restrict__ Sp,
                          const float* __restrict__ y_c, const float* __restrict__ l,
                          _Float16* __restrict__ zout) {
  const int mtile = blockIdx.y * (blockDim.x >> 5) + (threadIdx.x >> 5);
  if (mtile >= MT_B) return;
  const int lane = threadIdx.x & 31;
  const int half = lane >> 4;
  const int n    = blockIdx.x * 16 + (lane & 15);
  const int nb   = blockIdx.z;
  const _Float16* A    = zin + (size_t)nb * PPAD * 256;
  const _Float16* Bp_n = Sp + (size_t)nb * 65536 + (size_t)blockIdx.x * 4096;
  v8f acc = {};
  acc = wmma_rows(A, 256, mtile * 16, Bp_n, 8, acc);
#pragma unroll
  for (int r = 0; r < 8; ++r) {
    const size_t m   = (size_t)nb * PPAD + (size_t)mtile * 16 + r + 8 * half;
    const size_t idx = m * 256 + n;
    zout[idx] = (_Float16)softf(acc[r] + y_c[idx], l[idx]);
  }
}

// xp = Wg * clip(z @ Dc^T, 0, 1)
__global__ void gemm_recon(const _Float16* __restrict__ z, const _Float16* __restrict__ DcTp,
                           const float* __restrict__ Wg, float* __restrict__ xp) {
  const int mtile = blockIdx.y * (blockDim.x >> 5) + (threadIdx.x >> 5);
  if (mtile >= MT_B) return;
  const int lane = threadIdx.x & 31;
  const int half = lane >> 4;
  const int n    = blockIdx.x * 16 + (lane & 15);
  const int nb   = blockIdx.z;
  const _Float16* A    = z + (size_t)nb * PPAD * 256;
  const _Float16* Bp_n = DcTp + (size_t)nb * 16384 + (size_t)blockIdx.x * 4096;
  v8f acc = {};
  acc = wmma_rows(A, 256, mtile * 16, Bp_n, 8, acc);
#pragma unroll
  for (int r = 0; r < 8; ++r) {
    const size_t m   = (size_t)nb * PPAD + (size_t)mtile * 16 + r + 8 * half;
    const size_t idx = m * 64 + n;
    const float v = fminf(fmaxf(acc[r], 0.f), 1.f);
    xp[idx] = Wg[idx] * v;
  }
}

// ---------------------------------------------------------------- dictionary
__global__ void dict_bcast(const float* __restrict__ Dict, float* __restrict__ Dc) {
  const int tid = blockIdx.x * blockDim.x + threadIdx.x; // N*64*144
  if (tid >= NBATCH * 64 * 144) return;
  const int n = tid / (64 * 144);
  const int rem = tid % (64 * 144);
  const int d = rem / 144, a = rem % 144;
  Dc[((size_t)n * 64 + d) * 256 + a] = Dict[d * 144 + a];
}

// S = I - Dc^T Dc / c  (per batch, 256x256, sum over 64)
__global__ void compute_S(const float* __restrict__ Dc, float* __restrict__ S,
                          const float* __restrict__ c_ptr) {
  const int tid = blockIdx.x * blockDim.x + threadIdx.x; // N*65536
  if (tid >= NBATCH * 65536) return;
  const int n = tid >> 16;
  const int ij = tid & 65535;
  const int i = ij >> 8, j = ij & 255;
  const float* D = Dc + (size_t)n * 64 * 256;
  float s = 0.f;
#pragma unroll 4
  for (int d = 0; d < 64; ++d) s += D[d * 256 + i] * D[d * 256 + j];
  S[tid] = ((i == j) ? 1.f : 0.f) - s / c_ptr[0];
}

// CBAM on normalized sdict, writes adaptive atoms into Dc[:, :, 144:256]
__global__ void __launch_bounds__(128)
cbam_k(const float* __restrict__ sdraw,
       const float* __restrict__ ca1w, const float* __restrict__ ca1b,
       const float* __restrict__ ca2w, const float* __restrict__ ca2b,
       const float* __restrict__ saw, const float* __restrict__ sab,
       float* __restrict__ Dc) {
  __shared__ float sd[112][64];
  __shared__ float avg[64], mxv[64], ca[64];
  __shared__ float havg[128], hmx[128];
  __shared__ float s0[112], s1[112], sa[112];
  const int n = blockIdx.x;
  const int t = threadIdx.x;
  if (t < 112) {  // unit-norm rows (axis=-1, 64 dims)
    const float* row = sdraw + ((size_t)n * 112 + t) * 64;
    float s = 0.f;
    for (int d = 0; d < 64; ++d) { float v = row[d]; s += v * v; }
    const float inv = 1.f / fmaxf(sqrtf(s), 1e-12f);
    for (int d = 0; d < 64; ++d) sd[t][d] = row[d] * inv;
  }
  __syncthreads();
  if (t < 64) {   // channel avg/max over 112 spatial positions
    float a = 0.f, m = -1e30f;
    for (int q = 0; q < 112; ++q) { float v = sd[q][t]; a += v; m = fmaxf(m, v); }
    avg[t] = a * (1.f / 112.f); mxv[t] = m;
  }
  __syncthreads();
  {               // shared MLP hidden layer (64->128), both descriptors
    float ha = ca1b[t], hm = ca1b[t];
    for (int d = 0; d < 64; ++d) { ha += avg[d] * ca1w[d * 128 + t]; hm += mxv[d] * ca1w[d * 128 + t]; }
    havg[t] = fmaxf(ha, 0.f); hmx[t] = fmaxf(hm, 0.f);
  }
  __syncthreads();
  if (t < 64) {   // 128->64, mlp(avg)+mlp(max) includes bias twice
    float o = 2.f * ca2b[t];
    for (int k = 0; k < 128; ++k) o += (havg[k] + hmx[k]) * ca2w[k * 64 + t];
    ca[t] = 1.f / (1.f + expf(-o));
  }
  __syncthreads();
  if (t < 112) {  // spatial stats after channel attention
    float a = 0.f, m = -1e30f;
    for (int d = 0; d < 64; ++d) { float v = sd[t][d] * ca[d]; a += v; m = fmaxf(m, v); }
    s0[t] = a * (1.f / 64.f); s1[t] = m;
  }
  __syncthreads();
  if (t < 112) {  // 7x7 SAME conv over [mean, max] on 8x14 grid
    const int y = t / 14, x = t % 14;
    float accv = sab[0];
    for (int ky = 0; ky < 7; ++ky) {
      const int yy = y + ky - 3; if (yy < 0 || yy >= 8) continue;
      for (int kx = 0; kx < 7; ++kx) {
        const int xx = x + kx - 3; if (xx < 0 || xx >= 14) continue;
        const int qq = yy * 14 + xx;
        accv += s0[qq] * saw[ky * 7 + kx] + s1[qq] * saw[49 + ky * 7 + kx];
      }
    }
    sa[t] = 1.f / (1.f + expf(-accv));
  }
  __syncthreads();
  for (int e = t; e < 64 * 112; e += 128) {
    const int d = e / 112, q = e % 112;
    Dc[((size_t)n * 64 + d) * 256 + 144 + q] = sd[q][d] * ca[d] * sa[q];
  }
}

// ---------------------------------------------------------------- fold (gather)
__global__ void fold_k(const float* __restrict__ xp, const float* __restrict__ Wg,
                       float* __restrict__ out) {
  const int tid = blockIdx.x * blockDim.x + threadIdx.x; // N*128*128
  if (tid >= NBATCH * 16384) return;
  const int n = tid >> 14;
  const int pix = tid & 16383;
  const int yy = pix >> 7, xx = pix & 127;
  float num = 0.f, den = 0.f;
  const int i0 = (yy > 120) ? (yy - 120) : 0, i1 = (yy < 7) ? yy : 7;
  const int j0 = (xx > 120) ? (xx - 120) : 0, j1 = (xx < 7) ? xx : 7;
  for (int i = i0; i <= i1; ++i) {
    const int ho = yy - i;
    for (int j = j0; j <= j1; ++j) {
      const int wo = xx - j;
      const size_t idx = ((size_t)n * PPAD + (size_t)ho * 121 + wo) * 64 + i * 8 + j;
      num += xp[idx];
      den += Wg[idx];
    }
  }
  out[tid] = num / den;
}

// ---------------------------------------------------------------- launch
extern "C" void kernel_launch(void* const* d_in, const int* in_sizes, int n_in,
                              void* d_out, int out_size, void* d_ws, size_t ws_size,
                              hipStream_t stream) {
  (void)in_sizes; (void)n_in; (void)out_size; (void)ws_size;
  const float* x     = (const float*)d_in[0];
  const float* ca1w  = (const float*)d_in[31];
  const float* ca1b  = (const float*)d_in[32];
  const float* ca2w  = (const float*)d_in[33];
  const float* ca2b  = (const float*)d_in[34];
  const float* saw   = (const float*)d_in[35];
  const float* sab   = (const float*)d_in[36];
  const float* Dict  = (const float*)d_in[37];
  const float* c_ptr = (const float*)d_in[38];

  char* ws = (char*)d_ws;
  size_t cur = 0;
  auto take = [&](size_t bytes) -> void* {
    void* p = ws + cur;
    cur = (cur + bytes + 255) & ~(size_t)255;
    return p;
  };
  _Float16* unf_h = (_Float16*)take((size_t)MALL * 64 * 2);
  _Float16* ue_h  = (_Float16*)take(448 * 64 * 2);
  _Float16* sd1   = (_Float16*)take(448 * 256 * 2);
  _Float16* sd2   = (_Float16*)take(448 * 256 * 2);
  float*    sdraw = (float*)take(448 * 64 * 4);
  float*    Dc    = (float*)take((size_t)NBATCH * 64 * 256 * 4);
  float*    S     = (float*)take((size_t)NBATCH * 256 * 256 * 4);
  _Float16* Dcp   = (_Float16*)take((size_t)NBATCH * 64 * 256 * 2);
  _Float16* DcTp  = (_Float16*)take((size_t)NBATCH * 256 * 64 * 2);
  _Float16* Sp    = (_Float16*)take((size_t)NBATCH * 256 * 256 * 2);
  _Float16* ibufA = (_Float16*)take((size_t)MALL * 256 * 2);
  _Float16* ibufB = (_Float16*)take((size_t)MALL * 256 * 2);
  float*    lbuf  = (float*)take((size_t)MALL * 256 * 4);
  float*    y_c   = (float*)take((size_t)MALL * 256 * 4);
  _Float16* zA    = (_Float16*)take((size_t)MALL * 256 * 2);
  _Float16* zB    = (_Float16*)take((size_t)MALL * 256 * 2);
  float*    Wg    = (float*)take((size_t)MALL * 64 * 4);
  float*    xp    = (float*)take((size_t)MALL * 64 * 4);

  // 15 GEMM weights: {d_in index of W, K, Nout}; bias at index+1
  struct WInfo { int idx, K, N; };
  const WInfo wi[15] = {
      {1, 64, 128},  {3, 128, 256},  {5, 256, 128},  {7, 128, 64},    // ls1-4
      {9, 64, 128},  {11, 128, 256}, {13, 256, 128}, {15, 128, 112},  // l1-4
      {17, 64, 128}, {19, 128, 256}, {21, 256, 128}, {23, 128, 144},  // pd1-4
      {25, 64, 256}, {27, 256, 128}, {29, 128, 64},                   // w1-3
  };
  _Float16* wp[15];
  const float* wb[15];
  for (int i = 0; i < 15; ++i) {
    wp[i] = (_Float16*)take((size_t)wi[i].K * wi[i].N * 2);
    wb[i] = (const float*)d_in[wi[i].idx + 1];
  }

  // --- im2col + weight packing
  unfold_k<<<dim3(((size_t)MALL * 64 + 255) / 256), 256, 0, stream>>>(x, unf_h);
  ue_k<<<dim3((448 * 64 + 255) / 256), 256, 0, stream>>>(x, ue_h);
  for (int i = 0; i < 15; ++i) {
    const int tot = wi[i].K * wi[i].N;
    pack_b<<<dim3((tot + 255) / 256, 1, 1), 256, 0, stream>>>(
        (const float*)d_in[wi[i].idx], wp[i], wi[i].K, wi[i].N, wi[i].N, 1, 0, 0);
  }

  // --- adaptive dictionary: ls1..ls4 on 448 rows (28 tiles)
  gemm_mlp<<<dim3(8, 4), 256, 0, stream>>>(ue_h, wp[0], wb[0], 28, 2, sd1, nullptr, 128, 0, 0, nullptr);
  gemm_mlp<<<dim3(16, 4), 256, 0, stream>>>(sd1, wp[1], wb[1], 28, 4, sd2, nullptr, 256, 0, 0, nullptr);
  gemm_mlp<<<dim3(8, 4), 256, 0, stream>>>(sd2, wp[2], wb[2], 28, 8, sd1, nullptr, 128, 0, 0, nullptr);
  gemm_mlp<<<dim3(4, 4), 256, 0, stream>>>(sd1, wp[3], wb[3], 28, 4, nullptr, sdraw, 64, 0, 1, nullptr);
  cbam_k<<<dim3(NBATCH), 128, 0, stream>>>(sdraw, ca1w, ca1b, ca2w, ca2b, saw, sab, Dc);
  dict_bcast<<<dim3((NBATCH * 64 * 144 + 255) / 256), 256, 0, stream>>>(Dict, Dc);
  compute_S<<<dim3(NBATCH * 65536 / 256), 256, 0, stream>>>(Dc, S, c_ptr);
  pack_b<<<dim3(64 * 256 / 256, 1, NBATCH), 256, 0, stream>>>(Dc, Dcp, 64, 256, 256, 1, 64 * 256, 64 * 256);
  pack_b<<<dim3(256 * 64 / 256, 1, NBATCH), 256, 0, stream>>>(Dc, DcTp, 256, 64, 1, 256, 64 * 256, 256 * 64);
  pack_b<<<dim3(65536 / 256, 1, NBATCH), 256, 0, stream>>>(S, Sp, 256, 256, 256, 1, 65536, 65536);

  // --- big MLPs over all 58624 patch rows
  const dim3 gy(0, 458);  // 458*8 waves = 3664 m-tiles
  // lam: 64->128->256->128->112, result -> lbuf[:,144:256] / c
  gemm_mlp<<<dim3(8, 458), 256, 0, stream>>>(unf_h, wp[4], wb[4], MT_ALL, 2, ibufA, nullptr, 128, 0, 0, nullptr);
  gemm_mlp<<<dim3(16, 458), 256, 0, stream>>>(ibufA, wp[5], wb[5], MT_ALL, 4, ibufB, nullptr, 256, 0, 0, nullptr);
  gemm_mlp<<<dim3(8, 458), 256, 0, stream>>>(ibufB, wp[6], wb[6], MT_ALL, 8, ibufA, nullptr, 128, 0, 0, nullptr);
  gemm_mlp<<<dim3(7, 458), 256, 0, stream>>>(ibufA, wp[7], wb[7], MT_ALL, 4, nullptr, lbuf, 256, 144, 2, c_ptr);
  // lpd: 64->128->256->128->144, result -> lbuf[:,0:144] / c
  gemm_mlp<<<dim3(8, 458), 256, 0, stream>>>(unf_h, wp[8], wb[8], MT_ALL, 2, ibufA, nullptr, 128, 0, 0, nullptr);
  gemm_mlp<<<dim3(16, 458), 256, 0, stream>>>(ibufA, wp[9], wb[9], MT_ALL, 4, ibufB, nullptr, 256, 0, 0, nullptr);
  gemm_mlp<<<dim3(8, 458), 256, 0, stream>>>(ibufB, wp[10], wb[10], MT_ALL, 8, ibufA, nullptr, 128, 0, 0, nullptr);
  gemm_mlp<<<dim3(9, 458), 256, 0, stream>>>(ibufA, wp[11], wb[11], MT_ALL, 4, nullptr, lbuf, 256, 0, 2, c_ptr);
  // Wg: 64->256->128->64 sigmoid
  gemm_mlp<<<dim3(16, 458), 256, 0, stream>>>(unf_h, wp[12], wb[12], MT_ALL, 2, ibufA, nullptr, 256, 0, 0, nullptr);
  gemm_mlp<<<dim3(8, 458), 256, 0, stream>>>(ibufA, wp[13], wb[13], MT_ALL, 8, ibufB, nullptr, 128, 0, 0, nullptr);
  gemm_mlp<<<dim3(4, 458), 256, 0, stream>>>(ibufB, wp[14], wb[14], MT_ALL, 4, nullptr, Wg, 64, 0, 3, nullptr);

  // --- ISTA
  gemm_y<<<dim3(16, 115, NBATCH), 256, 0, stream>>>(unf_h, Dcp, lbuf, y_c, zA, c_ptr);
  _Float16* zi = zA;
  _Float16* zo = zB;
  for (int t = 0; t < 5; ++t) {
    gemm_ista<<<dim3(16, 115, NBATCH), 256, 0, stream>>>(zi, Sp, y_c, lbuf, zo);
    _Float16* tmp = zi; zi = zo; zo = tmp;
  }

  // --- reconstruction + fold
  gemm_recon<<<dim3(4, 115, NBATCH), 256, 0, stream>>>(zi, DcTp, Wg, xp);
  fold_k<<<dim3(NBATCH * 16384 / 256), 256, 0, stream>>>(xp, Wg, (float*)d_out);
}
